// Adapter_63522566307981
// MI455X (gfx1250) — compile-verified, matrix operand order checked by reading, not run on hardware
//
#include <hip/hip_runtime.h>
#include <math.h>

typedef _Float16 v16h __attribute__((ext_vector_type(16)));
typedef _Float16 v8h  __attribute__((ext_vector_type(8)));
typedef float    v8f  __attribute__((ext_vector_type(8)));

#define CIN 256
#define HID 1024
#define PIX 12544   /* 112*112 */
#define PT  64      /* pixels per workgroup */
#define NB  8

union V16 {
    v16h v;
    v8h  h[2];
};
union P8 {
    v8h v;
    _Float16 e[8];
};

// tanh-form GELU: one v_exp_f32 + one v_rcp_f32, ~3e-4 from exact erf GELU.
__device__ __forceinline__ float gelu_f(float x) {
    float z = 0.7978845608028654f * (x + 0.044715f * x * x * x);
    // tanh(z) = 1 - 2/(exp2(z*2/ln2)+1)
    float e = __builtin_amdgcn_exp2f(z * 2.8853900817779268f);
    float t = 1.0f - 2.0f * __builtin_amdgcn_rcpf(e + 1.0f);
    return 0.5f * x * (1.0f + t);
}

// Convert fp32 weights to f16 once; they stay resident in L2 (192MB) after.
__global__ void prep_weights(const float* __restrict__ w1,
                             const float* __restrict__ w2,
                             _Float16* __restrict__ w1h,
                             _Float16* __restrict__ w2h) {
    int i = blockIdx.x * blockDim.x + threadIdx.x;   // 0 .. 262143
    w1h[i] = (_Float16)w1[i];
    w2h[i] = (_Float16)w2[i];
}

__global__ __launch_bounds__(256, 4)
void convnext_block(const float* __restrict__ x,
                    const float* __restrict__ lnw,
                    const float* __restrict__ lnb,
                    const _Float16* __restrict__ w1h,
                    const float* __restrict__ b1,
                    const _Float16* __restrict__ w2h,
                    const float* __restrict__ b2,
                    float* __restrict__ out) {
    // B-operand swizzled tiles: tile (kb, pb) = 32 lanes x 16 halfs, lane-contiguous.
    __shared__ _Float16 XnL[8 * 4 * 32 * 16];   // 32 KB (K=256, 64 pixels)
    __shared__ _Float16 H1L[4 * 4 * 32 * 16];   // 16 KB (K-chunk=128, 64 pixels)
    __shared__ float redS[4 * 64];
    __shared__ float redQ[4 * 64];

    const int tid  = threadIdx.x;
    const int lane = tid & 31;
    const int w    = tid >> 5;          // wave id 0..7
    const int b    = blockIdx.x / (PIX / PT);
    const int p0   = (blockIdx.x % (PIX / PT)) * PT;

    const float* xb = x   + (size_t)b * CIN * PIX + p0;
    float*       ob = out + (size_t)b * CIN * PIX + p0;

    // ---------------- LayerNorm over channels, per pixel ----------------
    const int pl = tid & 63;            // pixel within tile
    const int q  = tid >> 6;            // channel quarter
    float s = 0.0f, ss = 0.0f;
    for (int c = q * 64; c < q * 64 + 64; ++c) {
        float v = xb[c * PIX + pl];     // coalesced: 64 lanes x contiguous pixels
        s += v; ss += v * v;
    }
    redS[q * 64 + pl] = s;
    redQ[q * 64 + pl] = ss;
    __syncthreads();
    float sum = redS[pl] + redS[64 + pl] + redS[128 + pl] + redS[192 + pl];
    float sq  = redQ[pl] + redQ[64 + pl] + redQ[128 + pl] + redQ[192 + pl];
    float mean = sum * (1.0f / 256.0f);
    float var  = sq * (1.0f / 256.0f) - mean * mean;
    float rstd = __builtin_amdgcn_rsqf(var + 1e-6f);

    const int pb_l = pl >> 4;
    const int nl   = pl & 15;
    #pragma unroll
    for (int g = 0; g < 8; ++g) {                    // 8 channels per b128 store
        const int c0 = q * 64 + g * 8;
        P8 pk;
        #pragma unroll
        for (int j = 0; j < 8; ++j) {
            int c = c0 + j;
            float v  = xb[c * PIX + pl];             // L2/L0 hot re-read
            float hn = (v - mean) * rstd;
            pk.e[j] = (_Float16)(lnw[c] * hn + lnb[c]);
        }
        // swizzle: kl0 = c0&31 is a multiple of 8 -> contiguous slot run
        int kb    = c0 >> 5;
        int kl0   = c0 & 31;
        int half  = (kl0 >> 3) & 1;
        int slot0 = ((kl0 >> 4) << 3);
        int laneD = half * 16 + nl;
        *(v8h*)&XnL[(((kb * 4 + pb_l) * 32) + laneD) * 16 + slot0] = pk.v;
    }
    __syncthreads();

    // ---------------- fused GEMM1 -> GELU -> GEMM2 ----------------
    const int aRow = lane & 15;   // tile row held by this lane (A operand)
    const int seg  = lane >> 4;   // which K-halves this lane holds
    const int mAdd = (lane >> 4) * 8;   // D layout: lanes 16-31 hold M=r+8

    v8f acc2[2][4] = {};          // persistent GEMM2 accumulators (64 VGPRs)

    for (int chunk = 0; chunk < 8; ++chunk) {
        // ---- GEMM1: this wave computes H rows [chunk*128 + w*16, +16) ----
        v8f acc1[4] = {};
        const int hbase = chunk * 128 + w * 16 + aRow;
        const _Float16* arow = w1h + hbase * CIN;
        if (chunk + 1 < 8) {      // warm next chunk's weight rows into cache
            __builtin_prefetch(arow + 128 * CIN, 0, 2);
            __builtin_prefetch(w2h + (w * 32 + aRow) * HID + (chunk + 1) * 128, 0, 2);
        }
        #pragma unroll
        for (int ks = 0; ks < 8; ++ks) {            // K = 256 in steps of 32
            V16 A;
            A.h[0] = *(const v8h*)(arow + ks * 32 + seg * 8);
            A.h[1] = *(const v8h*)(arow + ks * 32 + 16 + seg * 8);
            #pragma unroll
            for (int pb = 0; pb < 4; ++pb) {
                const v16h Bm = *(const v16h*)&XnL[((ks * 4 + pb) * 32 + lane) * 16];
                acc1[pb] = __builtin_amdgcn_wmma_f32_16x16x32_f16(
                    false, A.v, false, Bm, (short)0, acc1[pb], false, false);
            }
        }

        // bias vector for this wave's 16 H-rows (2 x b128 loads)
        float b1v[8];
        {
            const float* p = b1 + chunk * 128 + w * 16 + mAdd;
            float4 lo = *(const float4*)p;
            float4 hi = *(const float4*)(p + 4);
            b1v[0] = lo.x; b1v[1] = lo.y; b1v[2] = lo.z; b1v[3] = lo.w;
            b1v[4] = hi.x; b1v[5] = hi.y; b1v[6] = hi.z; b1v[7] = hi.w;
        }

        __syncthreads();   // previous chunk's H1 fully consumed

        // ---- bias + GELU, pack 8 f16, single ds_store_b128 per pixel-tile ----
        // Swizzle property: dest lane == own lane, slots (w&1)*8 .. +7.
        {
            const int kb2   = w >> 1;
            const int slot0 = (w & 1) * 8;
            #pragma unroll
            for (int pb = 0; pb < 4; ++pb) {
                P8 pk;
                #pragma unroll
                for (int r = 0; r < 8; ++r) {
                    pk.e[r] = (_Float16)gelu_f(acc1[pb][r] + b1v[r]);
                }
                *(v8h*)&H1L[((kb2 * 4 + pb) * 32 + lane) * 16 + slot0] = pk.v;
            }
        }
        __syncthreads();

        // ---- GEMM2 partial: K range [chunk*128, +128) ----
        #pragma unroll
        for (int ks2 = 0; ks2 < 4; ++ks2) {
            #pragma unroll
            for (int ct = 0; ct < 2; ++ct) {
                const int c0 = w * 32 + ct * 16;
                const _Float16* arow2 =
                    w2h + (c0 + aRow) * HID + chunk * 128 + ks2 * 32;
                V16 A;
                A.h[0] = *(const v8h*)(arow2 + seg * 8);
                A.h[1] = *(const v8h*)(arow2 + 16 + seg * 8);
                #pragma unroll
                for (int pb = 0; pb < 4; ++pb) {
                    const v16h Bm = *(const v16h*)&H1L[((ks2 * 4 + pb) * 32 + lane) * 16];
                    acc2[ct][pb] = __builtin_amdgcn_wmma_f32_16x16x32_f16(
                        false, A.v, false, Bm, (short)0, acc2[ct][pb], false, false);
                }
            }
        }
    }

    // ---------------- epilogue: +b2, +residual, GELU, store ----------------
    {
        const int n = lane & 15;
        #pragma unroll
        for (int ct = 0; ct < 2; ++ct) {
            float b2v[8];
            {
                const float* p = b2 + w * 32 + ct * 16 + mAdd;
                float4 lo = *(const float4*)p;
                float4 hi = *(const float4*)(p + 4);
                b2v[0] = lo.x; b2v[1] = lo.y; b2v[2] = lo.z; b2v[3] = lo.w;
                b2v[4] = hi.x; b2v[5] = hi.y; b2v[6] = hi.z; b2v[7] = hi.w;
            }
            #pragma unroll
            for (int pb = 0; pb < 4; ++pb) {
                #pragma unroll
                for (int r = 0; r < 8; ++r) {
                    int c = w * 32 + ct * 16 + r + mAdd;
                    int p = pb * 16 + n;
                    float xv  = xb[c * PIX + p];
                    float val = acc2[ct][pb][r] + b2v[r] + xv;
                    ob[c * PIX + p] = gelu_f(val);
                }
            }
        }
    }
}

extern "C" void kernel_launch(void* const* d_in, const int* in_sizes, int n_in,
                              void* d_out, int out_size, void* d_ws, size_t ws_size,
                              hipStream_t stream) {
    const float* x   = (const float*)d_in[0];
    const float* lnw = (const float*)d_in[1];
    const float* lnb = (const float*)d_in[2];
    const float* w1  = (const float*)d_in[3];
    const float* b1  = (const float*)d_in[4];
    const float* w2  = (const float*)d_in[5];
    const float* b2  = (const float*)d_in[6];
    float* out = (float*)d_out;

    _Float16* w1h = (_Float16*)d_ws;
    _Float16* w2h = w1h + (size_t)HID * CIN;

    prep_weights<<<(HID * CIN) / 256, 256, 0, stream>>>(w1, w2, w1h, w2h);
    convnext_block<<<NB * (PIX / PT), 256, 0, stream>>>(x, lnw, lnb, w1h, b1,
                                                        w2h, b2, out);
}
